// TriadicAttention_23167053594752
// MI455X (gfx1250) — matrix operand order repaired and checked by reading, not compiled, not run on hardware
//
#include <hip/hip_runtime.h>

// Problem constants (reference: B=2, T=2048, D=1024, H=16, DH=64)
#define BB 2
#define TT 2048
#define DD 1024
#define HH 16
#define DH 64
#define MTOT (BB * TT)          // 4096
#define NQKV (HH * DH)          // 1024

typedef __attribute__((ext_vector_type(16))) _Float16 v16h;
typedef __attribute__((ext_vector_type(8)))  _Float16 v8h;
typedef __attribute__((ext_vector_type(8)))  float    v8f;

// ---------------------------------------------------------------------------
// Fragment loaders for V_WMMA_F32_16X16X32_F16 (wave32)
// A (16x32, MxK, 16-bit): lane L -> row M=L&15, 16 contiguous K values
//   starting at (L>>4)*16 (same k order as B).
// B (32x16, KxN, 16-bit): lane L -> col N=L&15, 16 contiguous K values
//   starting at (L>>4)*16.  Bt is stored transposed [N,K] row-major so the
//   whole per-lane fragment is one contiguous 32-byte load.
// ---------------------------------------------------------------------------
__device__ __forceinline__ v16h load_a_frag(const _Float16* A, int lda, int m0, int k0) {
    int lane = threadIdx.x & 31;
    int row  = lane & 15;
    int kb   = (lane >> 4) << 4;  // 0 or 16
    const _Float16* p = A + (size_t)(m0 + row) * lda + k0;
    union { v16h v; v8h h[2]; } u;
    u.h[0] = *(const v8h*)(p + kb);
    u.h[1] = *(const v8h*)(p + 8 + kb);
    return u.v;
}

__device__ __forceinline__ v16h load_b_frag(const _Float16* Bt, int ldb, int n0, int k0) {
    int lane = threadIdx.x & 31;
    int col  = lane & 15;
    int kb   = (lane >> 4) << 4;  // 0 or 16
    return *(const v16h*)(Bt + (size_t)(n0 + col) * ldb + k0 + kb);
}

// ---------------------------------------------------------------------------
// 64x32 register-blocked WMMA accumulation: acc += A[tm:tm+64] @ Bt[tn:tn+32]^T
// 8 independent WMMAs per 32-wide K-step.  Register budget ~150 VGPRs
// (8 f32 accumulators = 64, 6 fragments = 48, addressing) -> no spills.
// ---------------------------------------------------------------------------
__device__ __forceinline__ void wmma_block_64x32(const _Float16* __restrict__ A,
                                                 const _Float16* __restrict__ Bt,
                                                 int lda, int ldb, int K,
                                                 int tm, int tn, v8f (&acc)[4][2]) {
    for (int k0 = 0; k0 < K; k0 += 32) {
        if (k0 + 32 < K) {
            __builtin_prefetch(A + (size_t)tm * lda + k0 + 32, 0, 1);
            __builtin_prefetch(Bt + (size_t)tn * ldb + k0 + 32, 0, 1);
        }
        v16h af[4], bf[2];
        #pragma unroll
        for (int i = 0; i < 4; ++i) af[i] = load_a_frag(A, lda, tm + (i << 4), k0);
        #pragma unroll
        for (int j = 0; j < 2; ++j) bf[j] = load_b_frag(Bt, ldb, tn + (j << 4), k0);
        #pragma unroll
        for (int i = 0; i < 4; ++i)
            #pragma unroll
            for (int j = 0; j < 2; ++j)
                acc[i][j] = __builtin_amdgcn_wmma_f32_16x16x32_f16(
                    false, af[i], false, bf[j], (short)0, acc[i][j], false, false);
    }
}

// ---------------------------------------------------------------------------
// Elementwise conversion / transpose prep kernels
// ---------------------------------------------------------------------------
__global__ void cvt_f32_to_f16_kernel(const float* __restrict__ src,
                                      _Float16* __restrict__ dst, int n) {
    int i = blockIdx.x * blockDim.x + threadIdx.x;
    if (i < n) dst[i] = (_Float16)src[i];
}

// Wt[n*K + k] = W[k*N + n]   (W is [K,N] row-major f32; Wt is [N,K] f16)
__global__ void transpose_to_f16_kernel(const float* __restrict__ W,
                                        _Float16* __restrict__ Wt, int K, int N) {
    int i = blockIdx.x * blockDim.x + threadIdx.x;
    if (i < K * N) {
        int n = i / K;
        int k = i - n * K;
        Wt[i] = (_Float16)W[(size_t)k * N + n];
    }
}

// ---------------------------------------------------------------------------
// Fused q/k/v projection: blockIdx.y selects weight + output handling.
//   proj 0: q = (x@Wq) * DH^-0.5  -> f16, [B,H,T,DH]
//   proj 1: k =  x@Wk             -> f16, [B,H,T,DH]
//   proj 2: v =  x@Wv             -> f32, [B,H,T,DH]
// ---------------------------------------------------------------------------
__global__ void qkv_gemm_kernel(const _Float16* __restrict__ xh,
                                const _Float16* __restrict__ WqT,
                                const _Float16* __restrict__ WkT,
                                const _Float16* __restrict__ WvT,
                                _Float16* __restrict__ qh,
                                _Float16* __restrict__ kh,
                                float* __restrict__ vf) {
    int proj = blockIdx.y;
    const _Float16* Bt = (proj == 0) ? WqT : ((proj == 1) ? WkT : WvT);

    const int tiles_n = NQKV >> 5;              // 32
    int bid = blockIdx.x;
    int tm = (bid / tiles_n) << 6;
    int tn = (bid % tiles_n) << 5;

    v8f acc[4][2];
    #pragma unroll
    for (int i = 0; i < 4; ++i)
        #pragma unroll
        for (int j = 0; j < 2; ++j) acc[i][j] = (v8f){};

    wmma_block_64x32(xh, Bt, DD, DD, DD, tm, tn, acc);

    float scale = (proj == 0) ? 0.125f : 1.0f;  // DH^-0.5 folded into q
    int lane = threadIdx.x & 31;
    int nloc = lane & 15;
    int mloc = (lane >> 4) << 3;
    #pragma unroll
    for (int i = 0; i < 4; ++i) {
        #pragma unroll
        for (int j = 0; j < 2; ++j) {
            #pragma unroll
            for (int r = 0; r < 8; ++r) {
                int m = tm + (i << 4) + mloc + r;
                int n = tn + (j << 4) + nloc;
                float val = acc[i][j][r] * scale;
                int b_ = m >> 11;              // T = 2048
                int t  = m & (TT - 1);
                int h  = n >> 6;               // DH = 64
                int d  = n & (DH - 1);
                size_t o = ((size_t)(b_ * HH + h) * TT + t) * DH + d;
                if (proj == 0)      qh[o] = (_Float16)val;
                else if (proj == 1) kh[o] = (_Float16)val;
                else                vf[o] = val;
            }
        }
    }
}

// ---------------------------------------------------------------------------
// Output projection: out = y @ Wo  (f16 in, f32 out, plain row-major [M,N])
// ---------------------------------------------------------------------------
__global__ void out_gemm_kernel(const _Float16* __restrict__ yh,
                                const _Float16* __restrict__ WoT,
                                float* __restrict__ out) {
    const int tiles_n = DD >> 5;                // 32
    int bid = blockIdx.x;
    int tm = (bid / tiles_n) << 6;
    int tn = (bid % tiles_n) << 5;

    v8f acc[4][2];
    #pragma unroll
    for (int i = 0; i < 4; ++i)
        #pragma unroll
        for (int j = 0; j < 2; ++j) acc[i][j] = (v8f){};

    wmma_block_64x32(yh, WoT, NQKV, NQKV, NQKV, tm, tn, acc);

    int lane = threadIdx.x & 31;
    int nloc = lane & 15;
    int mloc = (lane >> 4) << 3;
    #pragma unroll
    for (int i = 0; i < 4; ++i) {
        #pragma unroll
        for (int j = 0; j < 2; ++j) {
            #pragma unroll
            for (int r = 0; r < 8; ++r) {
                int m = tm + (i << 4) + mloc + r;
                int n = tn + (j << 4) + nloc;
                out[(size_t)m * DD + n] = acc[i][j][r];
            }
        }
    }
}

// ---------------------------------------------------------------------------
// Top-2 neighbor search per query row: scores = q @ k^T (diagonal masked).
// One wave per (b,h, 16-query tile).  q pre-scaled; scale irrelevant to argtop2.
// ---------------------------------------------------------------------------
__global__ void top2_kernel(const _Float16* __restrict__ qh,
                            const _Float16* __restrict__ kh,
                            int* __restrict__ idxj, int* __restrict__ idxk) {
    __shared__ float tile[16 * 17];
    int bid = blockIdx.x;
    int qt  = bid & ((TT / 16) - 1);   // 0..127
    int bh  = bid >> 7;                // 0..B*H-1
    const _Float16* Q  = qh + (size_t)bh * TT * DH;
    const _Float16* Kt = kh + (size_t)bh * TT * DH;
    int lane = threadIdx.x & 31;

    // A fragments for this query tile (DH=64 -> two K-steps), loaded once.
    v16h a0 = load_a_frag(Q, DH, qt << 4, 0);
    v16h a1 = load_a_frag(Q, DH, qt << 4, 32);

    float v1 = -__builtin_inff(), v2 = -__builtin_inff();
    int   i1 = 0, i2 = 0;
    int   qrow = (qt << 4) + lane;     // meaningful for lane < 16

    for (int st = 0; st < TT / 16; ++st) {
        v16h b0 = load_b_frag(Kt, DH, st << 4, 0);
        v16h b1 = load_b_frag(Kt, DH, st << 4, 32);
        v8f c = {};
        c = __builtin_amdgcn_wmma_f32_16x16x32_f16(false, a0, false, b0,
                                                   (short)0, c, false, false);
        c = __builtin_amdgcn_wmma_f32_16x16x32_f16(false, a1, false, b1,
                                                   (short)0, c, false, false);
        // Spill 16x16 f32 tile to LDS (lane = column, VGPR = row)
        int n  = lane & 15;
        int mb = (lane >> 4) << 3;
        #pragma unroll
        for (int r = 0; r < 8; ++r) tile[(mb + r) * 17 + n] = c[r];
        __syncthreads();
        if (lane < 16) {
            #pragma unroll
            for (int j = 0; j < 16; ++j) {
                int s = (st << 4) + j;
                if (s == qrow) continue;           // diagonal -> -inf
                float val = tile[lane * 17 + j];
                if (val > v1)      { v2 = v1; i2 = i1; v1 = val; i1 = s; }
                else if (val > v2) { v2 = val; i2 = s; }
            }
        }
        __syncthreads();
    }
    if (lane < 16) {
        int t = (qt << 4) + lane;
        idxj[(size_t)bh * TT + t] = i1;
        idxk[(size_t)bh * TT + t] = i2;
    }
}

// ---------------------------------------------------------------------------
// Soft median of {v, v_j, v_k}: exact median init + 2 Newton steps on
// Huberized L1 (tau = 1e-3), matching the reference elementwise math.
// ---------------------------------------------------------------------------
__device__ __forceinline__ float huber_g(float u, float x) {
    float r = u - x;
    float s = (r > 0.f) ? 1.f : ((r < 0.f) ? -1.f : 0.f);
    return (__builtin_fabsf(r) <= 1e-3f) ? r : 1e-3f * s;
}
__device__ __forceinline__ float huber_h(float u, float x) {
    return (__builtin_fabsf(u - x) <= 1e-3f) ? 1.f : 0.f;
}

__global__ void soft_median_kernel(const float* __restrict__ vf,
                                   const int* __restrict__ idxj,
                                   const int* __restrict__ idxk,
                                   _Float16* __restrict__ yh) {
    int i = blockIdx.x * blockDim.x + threadIdx.x;  // over B*H*T*DH
    if (i >= BB * HH * TT * DH) return;
    int d  = i & (DH - 1);
    int t  = (i >> 6) & (TT - 1);
    int bh = i >> 17;                               // TT*DH = 2^17
    float a  = vf[i];
    int tj = idxj[(size_t)bh * TT + t];
    int tk = idxk[(size_t)bh * TT + t];
    float b = vf[((size_t)bh * TT + tj) * DH + d];
    float c = vf[((size_t)bh * TT + tk) * DH + d];

    float u = fmaxf(fminf(a, b), fminf(fmaxf(a, b), c));  // median of 3
    #pragma unroll
    for (int it = 0; it < 2; ++it) {
        float g  = huber_g(u, a) + huber_g(u, b) + huber_g(u, c);
        float Hd = huber_h(u, a) + huber_h(u, b) + huber_h(u, c);
        float step = (Hd > 0.f) ? g / fmaxf(Hd, 1e-6f) : 0.f;
        u -= step;
    }
    // write as [B*T, H*DH] f16 for the final GEMM
    int b_ = bh >> 4, h = bh & (HH - 1);
    yh[((size_t)(b_ * TT + t)) * NQKV + h * DH + d] = (_Float16)u;
}

// ---------------------------------------------------------------------------
// Launch
// ---------------------------------------------------------------------------
extern "C" void kernel_launch(void* const* d_in, const int* in_sizes, int n_in,
                              void* d_out, int out_size, void* d_ws, size_t ws_size,
                              hipStream_t stream) {
    const float* x  = (const float*)d_in[0];
    const float* Wq = (const float*)d_in[1];
    const float* Wk = (const float*)d_in[2];
    const float* Wv = (const float*)d_in[3];
    const float* Wo = (const float*)d_in[4];
    float* out = (float*)d_out;

    char* ws = (char*)d_ws;
    size_t off = 0;
    auto alloc = [&](size_t bytes) -> void* {
        void* p = ws + off;
        off = (off + bytes + 255) & ~(size_t)255;
        return p;
    };

    const size_t nX   = (size_t)MTOT * DD;         // 4M
    const size_t nW   = (size_t)DD * NQKV;         // 1M per weight
    const size_t nQKV = (size_t)BB * HH * TT * DH; // 4M

    _Float16* xh  = (_Float16*)alloc(nX * 2);
    _Float16* WqT = (_Float16*)alloc(nW * 2);
    _Float16* WkT = (_Float16*)alloc(nW * 2);
    _Float16* WvT = (_Float16*)alloc(nW * 2);
    _Float16* WoT = (_Float16*)alloc(nW * 2);
    _Float16* qh  = (_Float16*)alloc(nQKV * 2);
    _Float16* kh  = (_Float16*)alloc(nQKV * 2);
    float*    vf  = (float*)   alloc(nQKV * 4);
    int*      ij  = (int*)     alloc((size_t)BB * HH * TT * 4);
    int*      ik  = (int*)     alloc((size_t)BB * HH * TT * 4);
    _Float16* yh  = (_Float16*)alloc(nX * 2);

    // 1) precision prep
    cvt_f32_to_f16_kernel<<<(int)((nX + 255) / 256), 256, 0, stream>>>(x, xh, (int)nX);
    transpose_to_f16_kernel<<<(int)((nW + 255) / 256), 256, 0, stream>>>(Wq, WqT, DD, NQKV);
    transpose_to_f16_kernel<<<(int)((nW + 255) / 256), 256, 0, stream>>>(Wk, WkT, DD, NQKV);
    transpose_to_f16_kernel<<<(int)((nW + 255) / 256), 256, 0, stream>>>(Wv, WvT, DD, NQKV);
    transpose_to_f16_kernel<<<(int)((nW + 255) / 256), 256, 0, stream>>>(Wo, WoT, NQKV, DD);

    // 2) fused q/k/v projections: 64x32 tile per wave, 3 projections in one grid
    const int tiles = (MTOT / 64) * (NQKV / 32);    // 2048
    qkv_gemm_kernel<<<dim3(tiles, 3), 32, 0, stream>>>(xh, WqT, WkT, WvT, qh, kh, vf);

    // 3) top-2 neighbors via WMMA score tiles
    top2_kernel<<<BB * HH * (TT / 16), 32, 0, stream>>>(qh, kh, ij, ik);

    // 4) elementwise soft median -> y (f16)
    soft_median_kernel<<<(int)((nQKV + 255) / 256), 256, 0, stream>>>(vf, ij, ik, yh);

    // 5) output projection -> f32 d_out
    out_gemm_kernel<<<tiles, 32, 0, stream>>>(yh, WoT, out);
}
